// Block_Head_15212774162755
// MI455X (gfx1250) — matrix operand order repaired and checked by reading, not compiled, not run on hardware
//
#include <hip/hip_runtime.h>

// ---------------------------------------------------------------------------
// Types for gfx1250 WMMA
// ---------------------------------------------------------------------------
typedef __attribute__((ext_vector_type(16))) __bf16 v16bf;
typedef __attribute__((ext_vector_type(8)))  float  v8f;
typedef __attribute__((ext_vector_type(4)))  unsigned int u32x4;

__device__ __forceinline__ unsigned short f2bf(float f) {
    unsigned int u = __float_as_uint(f);
    unsigned int r = u + 0x7FFFu + ((u >> 16) & 1u);
    return (unsigned short)(r >> 16);
}
__device__ __forceinline__ float silu(float x) { return x / (1.f + __expf(-x)); }

// ---------------------------------------------------------------------------
// bf16 GEMM: C(MxN) = A(MxK) @ Wt(KxN)   (Wt is the transposed weight, bf16)
// Block = 8 waves (4 in M x 2 in N) -> 128x128 block tile.
// Wave computes a 32x64 tile = 2x4 WMMA 16x16 tiles -> 8 v_wmma per k-step,
// 4 b128 A loads + 64 scalar 16-bit B loads per k-step (A loaded once per 4 N-tiles).
// A-fragment (16-bit A 16x32): lane<16 -> K[k..k+7],K[k+16..k+23];
//   lane>=16 -> K[k+8..k+15],K[k+24..k+31]; row m = lane&15.
// B-fragment (16-bit B 32x16): col n = lane&15, elements = K in the same order as A.
// ---------------------------------------------------------------------------
__global__ __launch_bounds__(256) void gemm_bf16_nt(
    const unsigned short* __restrict__ A,
    const unsigned short* __restrict__ Bt,
    float* __restrict__ C, int M, int N, int K)
{
    const int lane = threadIdx.x & 31;
    const int wave = threadIdx.x >> 5;
    const int wm = wave >> 1, wn = wave & 1;
    const int m0 = blockIdx.y * 128 + wm * 32;
    const int n0 = blockIdx.x * 128 + wn * 64;
    if (m0 >= M || n0 >= N) return;

    v8f acc[2][4] = {};
    const int r   = lane & 15;
    const int klo = (lane >> 4) * 8;

    const unsigned short* ap0 = A + (size_t)(m0 + r) * K + klo;
    const unsigned short* ap1 = ap0 + (size_t)16 * K;
    const unsigned short* bp  = Bt + (size_t)klo * N + n0 + r;
    const size_t bstep = (size_t)32 * N;

    for (int k = 0; k < K; k += 32) {
        v16bf a[2], b[4];
        *((u32x4*)&a[0])     = *(const u32x4*)(ap0);        // K: k+klo   .. +7
        *((u32x4*)&a[0] + 1) = *(const u32x4*)(ap0 + 16);   // K: k+16+klo.. +7
        *((u32x4*)&a[1])     = *(const u32x4*)(ap1);
        *((u32x4*)&a[1] + 1) = *(const u32x4*)(ap1 + 16);
#pragma unroll
        for (int nt = 0; nt < 4; ++nt)
            for (int s = 0; s < 16; ++s)                     // col n = lane&15, K in table order
                b[nt][s] = *(const __bf16*)(bp + (size_t)((s & 7) + (s >> 3) * 16) * N + nt * 16);

        if (k + 32 < K)                                      // next B panel row
            __builtin_prefetch(bp + bstep, 0, 3);            // -> global_prefetch_b8

#pragma unroll
        for (int mt = 0; mt < 2; ++mt)
#pragma unroll
            for (int nt = 0; nt < 4; ++nt)
                acc[mt][nt] = __builtin_amdgcn_wmma_f32_16x16x32_bf16(
                    false, a[mt], false, b[nt], (short)0, acc[mt][nt], false, false);

        ap0 += 32; ap1 += 32; bp += bstep;
    }

    // C/D layout: lane holds col n = lane&15, rows m = rr + (lane>=16 ? 8 : 0)
    const int moff = (lane >> 4) * 8;
#pragma unroll
    for (int mt = 0; mt < 2; ++mt) {
        float* crow = C + (size_t)(m0 + mt * 16 + moff) * N + n0 + r;
#pragma unroll
        for (int rr = 0; rr < 8; ++rr) {
#pragma unroll
            for (int nt = 0; nt < 4; ++nt)
                crow[nt * 16] = acc[mt][nt][rr];             // const 64B offsets
            crow += N;
        }
    }
}

// ---------------------------------------------------------------------------
// fp32 (NxK) weight -> bf16 (KxN) transpose-convert
// ---------------------------------------------------------------------------
__global__ void convT_bf16(const float* __restrict__ w, unsigned short* __restrict__ o,
                           int N, int K, long n)
{
    long i = (long)blockIdx.x * 256 + threadIdx.x;
    if (i >= n) return;
    int nn = (int)(i / K), k = (int)(i % K);
    o[(size_t)k * N + nn] = f2bf(w[i]);
}

// ---------------------------------------------------------------------------
// rmsnorm over D, fused weight scale, bf16 output.
// grid = (rows_per_batch R, B);  x row = base + b*bstride + r*D ; out row = b*R+r
// ---------------------------------------------------------------------------
__global__ __launch_bounds__(256) void rmsnorm_bf16(
    const float* __restrict__ x, const float* __restrict__ w,
    unsigned short* __restrict__ out, int D, long bstride)
{
    const float* xr = x + (size_t)blockIdx.y * bstride + (size_t)blockIdx.x * D;
    unsigned short* orow = out + ((size_t)blockIdx.y * gridDim.x + blockIdx.x) * D;
    __shared__ float red[256];
    float s = 0.f;
    for (int i = threadIdx.x; i < D; i += 256) { float v = xr[i]; s += v * v; }
    red[threadIdx.x] = s; __syncthreads();
    for (int off = 128; off > 0; off >>= 1) {
        if (threadIdx.x < off) red[threadIdx.x] += red[threadIdx.x + off];
        __syncthreads();
    }
    float scale = rsqrtf(red[0] / (float)D + 1e-6f);
    for (int i = threadIdx.x; i < D; i += 256)
        orow[i] = f2bf(xr[i] * scale * w[i]);
}

// ---------------------------------------------------------------------------
// depthwise conv (4 taps, causal) + bias + SiLU; reads xi = xz[:, :din]
// grid = (din/256, T, B)
// ---------------------------------------------------------------------------
__global__ __launch_bounds__(256) void conv_silu(
    const float* __restrict__ xz, const float* __restrict__ cw,
    const float* __restrict__ cb, float* __restrict__ xc,
    unsigned short* __restrict__ xcbf, int T, int din)
{
    const int d = blockIdx.x * 256 + threadIdx.x;
    const int t = blockIdx.y, b = blockIdx.z;
    float acc = cb[d];
#pragma unroll
    for (int k = 0; k < 4; ++k) {
        int ti = t - 3 + k;
        if (ti >= 0) acc += xz[(size_t)(b * T + ti) * (2 * din) + d] * cw[d * 4 + k];
    }
    float sv = silu(acc);
    size_t o = (size_t)(b * T + t) * din + d;
    xc[o] = sv; xcbf[o] = f2bf(sv);
}

__global__ void softplus_bias(float* __restrict__ dtf, const float* __restrict__ bdt,
                              int din, long n)
{
    long i = (long)blockIdx.x * 256 + threadIdx.x;
    if (i >= n) return;
    float v = dtf[i] + bdt[i % din];
    dtf[i] = (v > 20.f) ? v : log1pf(__expf(v));
}

__global__ void slice_bf16(const float* __restrict__ src, unsigned short* __restrict__ o,
                           int cols_in, int cols_out, long n)
{
    long i = (long)blockIdx.x * 256 + threadIdx.x;
    if (i >= n) return;
    long r = i / cols_out; int c = (int)(i % cols_out);
    o[i] = f2bf(src[(size_t)r * cols_in + c]);
}

__global__ void silu_mul_bf16(const float* __restrict__ g, const float* __restrict__ u,
                              unsigned short* __restrict__ h, long n)
{
    long i = (long)blockIdx.x * 256 + threadIdx.x;
    if (i >= n) return;
    h[i] = f2bf(silu(g[i]) * u[i]);
}

// out[b, rowbase + r, :] = x[b, rowbase + r, :] + y[b*R + r, :]
__global__ void residual_add(const float* __restrict__ x, const float* __restrict__ y,
                             float* __restrict__ out, int R, int D, int L, int rowbase)
{
    long idx = (long)blockIdx.x * 256 + threadIdx.x;
    int r = (int)(idx / D), d = (int)(idx % D);
    int b = blockIdx.y;
    size_t xo = ((size_t)b * L + rowbase + r) * D + d;
    out[xo] = x[xo] + y[((size_t)b * R + r) * D + d];
}

// ---------------------------------------------------------------------------
// Selective scan. grid = (din/64, B), block = 256 (4 state-groups x 64 chans).
// Each thread holds 32 h-states + 32 A values in VGPRs; B_t/C_t staged in LDS.
// Fused epilogue: ybf = (ys + u*Dp) * silu(z)  ->  bf16
// ---------------------------------------------------------------------------
__global__ __launch_bounds__(256) void mamba_scan(
    const float* __restrict__ dt, const float* __restrict__ u,
    const float* __restrict__ xdbl, const float* __restrict__ A_log,
    const float* __restrict__ Dp, const float* __restrict__ xz,
    unsigned short* __restrict__ ybf, int T, int din)
{
    const int tid = threadIdx.x;
    const int sg = tid & 3;          // state group: states [sg*32, sg*32+32)
    const int dl = tid >> 2;         // local channel 0..63
    const int d = blockIdx.x * 64 + dl;
    const int b = blockIdx.y;

    float h[32], Av[32];
#pragma unroll
    for (int i = 0; i < 32; ++i) {
        h[i] = 0.f;
        Av[i] = -__expf(A_log[(size_t)d * 128 + sg * 32 + i]);
    }
    const float Dd = Dp[d];

    __shared__ float sB[128];
    __shared__ float sC[128];

    for (int t = 0; t < T; ++t) {
        const size_t row = (size_t)(b * T + t);
        if (tid < 128)      sB[tid] = xdbl[row * 320 + 64 + tid];
        else                sC[tid - 128] = xdbl[row * 320 + 192 + (tid - 128)];
        __syncthreads();

        const float dtv = dt[row * din + d];
        const float uv  = u[row * din + d];
        const float du  = dtv * uv;
        float y = 0.f;
#pragma unroll
        for (int i = 0; i < 32; ++i) {
            const int s = sg * 32 + i;
            float dA = __expf(dtv * Av[i]);
            h[i] = h[i] * dA + du * sB[s];
            y += h[i] * sC[s];
        }
        y += __shfl_xor(y, 1, 32);
        y += __shfl_xor(y, 2, 32);
        if (sg == 0) {
            float zv = xz[row * (2 * din) + din + d];
            ybf[row * din + d] = f2bf((y + uv * Dd) * silu(zv));
        }
        __syncthreads();
    }
}

// ---------------------------------------------------------------------------
// Host orchestration
// ---------------------------------------------------------------------------
static inline int cdiv(int a, int b) { return (a + b - 1) / b; }

extern "C" void kernel_launch(void* const* d_in, const int* in_sizes, int n_in,
                              void* d_out, int out_size, void* d_ws, size_t ws_size,
                              hipStream_t stream) {
    (void)in_sizes; (void)n_in; (void)out_size; (void)ws_size;
    // shapes (fixed by setup_inputs)
    const int B = 2, T = 1024, Ntok = 512, D = 1024;
    const int din = 4096, dstate = 128, dtr = 64;
    const int L = 3 * T + 5 * Ntok;          // 5632
    const int Mm = B * T;                    // 2048
    const int Mt = 5 * B * Ntok;             // 5120
    (void)dstate;

    const float* x        = (const float*)d_in[0];
    const float* nm_w     = (const float*)d_in[3];
    const float* nt_w     = (const float*)d_in[4];
    const float* in_proj  = (const float*)d_in[5];   // (3, 2*din, D)
    const float* conv_w   = (const float*)d_in[6];   // (3, din, 4)
    const float* conv_b   = (const float*)d_in[7];   // (3, din)
    const float* x_proj   = (const float*)d_in[8];   // (3, 320, din)
    const float* dt_proj  = (const float*)d_in[9];   // (3, din, 64)
    const float* dt_b     = (const float*)d_in[10];  // (3, din)
    const float* A_log    = (const float*)d_in[11];  // (3, din, 128)
    const float* D_param  = (const float*)d_in[12];  // (3, din)
    const float* out_proj = (const float*)d_in[13];  // (3, D, din)
    const float* wg       = (const float*)d_in[14];  // (4096, D)
    const float* wu       = (const float*)d_in[15];
    const float* wd       = (const float*)d_in[16];  // (D, 4096)
    float* out = (float*)d_out;

    // workspace carve-up
    size_t off = 0;
    auto take = [&](size_t bytes) { size_t r = off; off = (off + bytes + 255) & ~(size_t)255; return r; };
    char* ws = (char*)d_ws;

    unsigned short* w_in  = (unsigned short*)(ws + take((size_t)3 * D * (2 * din) * 2)); // KxN per layer
    unsigned short* w_x   = (unsigned short*)(ws + take((size_t)3 * din * 320 * 2));
    unsigned short* w_dt  = (unsigned short*)(ws + take((size_t)3 * dtr * din * 2));
    unsigned short* w_out = (unsigned short*)(ws + take((size_t)3 * din * D * 2));
    unsigned short* w_gb  = (unsigned short*)(ws + take((size_t)D * 4096 * 2));
    unsigned short* w_ub  = (unsigned short*)(ws + take((size_t)D * 4096 * 2));
    unsigned short* w_db  = (unsigned short*)(ws + take((size_t)4096 * D * 2));
    unsigned short* nrm   = (unsigned short*)(ws + take((size_t)Mt * D * 2)); // big enough for both phases
    size_t actBase = off;

    // mamba-phase activation pointers (reused every layer)
    off = actBase;
    float*          xz    = (float*)(ws + take((size_t)Mm * 2 * din * 4));
    float*          xc    = (float*)(ws + take((size_t)Mm * din * 4));
    unsigned short* xcbf  = (unsigned short*)(ws + take((size_t)Mm * din * 2));
    float*          xdbl  = (float*)(ws + take((size_t)Mm * 320 * 4));
    unsigned short* dtin  = (unsigned short*)(ws + take((size_t)Mm * dtr * 2));
    float*          dtf   = (float*)(ws + take((size_t)Mm * din * 4));
    unsigned short* ybf   = (unsigned short*)(ws + take((size_t)Mm * din * 2));
    float*          po    = (float*)(ws + take((size_t)Mm * D * 4));

    // text-phase pointers (same region, used after mamba phase completes)
    off = actBase;
    float*          gbuf  = (float*)(ws + take((size_t)Mt * 4096 * 4));
    float*          ubuf  = (float*)(ws + take((size_t)Mt * 4096 * 4));
    unsigned short* hbf   = (unsigned short*)(ws + take((size_t)Mt * 4096 * 2));
    float*          tout  = (float*)(ws + take((size_t)Mt * D * 4));

    auto launch_convT = [&](const float* w, unsigned short* o, int N, int K) {
        long n = (long)N * K;
        convT_bf16<<<dim3((unsigned)((n + 255) / 256)), 256, 0, stream>>>(w, o, N, K, n);
    };
    auto launch_gemm = [&](const unsigned short* A, const unsigned short* Bt, float* C,
                           int M, int N, int K) {
        gemm_bf16_nt<<<dim3(cdiv(N, 128), cdiv(M, 128)), 256, 0, stream>>>(A, Bt, C, M, N, K);
    };

    // --- weight convert + transpose (bf16, KxN) ---
    for (int l = 0; l < 3; ++l) {
        launch_convT(in_proj  + (size_t)l * (2 * din) * D, w_in  + (size_t)l * D * (2 * din), 2 * din, D);
        launch_convT(x_proj   + (size_t)l * 320 * din,     w_x   + (size_t)l * din * 320,     320,     din);
        launch_convT(dt_proj  + (size_t)l * din * dtr,     w_dt  + (size_t)l * dtr * din,     din,     dtr);
        launch_convT(out_proj + (size_t)l * D * din,       w_out + (size_t)l * din * D,       D,       din);
    }
    launch_convT(wg, w_gb, 4096, D);
    launch_convT(wu, w_ub, 4096, D);
    launch_convT(wd, w_db, D, 4096);

    // --- motion: 3 mamba layers ---
    for (int l = 0; l < 3; ++l) {
        rmsnorm_bf16<<<dim3(T, B), 256, 0, stream>>>(
            x + (size_t)l * T * D, nm_w + (size_t)l * D, nrm, D, (long)L * D);
        launch_gemm(nrm, w_in + (size_t)l * D * (2 * din), xz, Mm, 2 * din, D);
        conv_silu<<<dim3(din / 256, T, B), 256, 0, stream>>>(
            xz, conv_w + (size_t)l * din * 4, conv_b + (size_t)l * din, xc, xcbf, T, din);
        launch_gemm(xcbf, w_x + (size_t)l * din * 320, xdbl, Mm, 320, din);
        {
            long n = (long)Mm * dtr;
            slice_bf16<<<dim3((unsigned)((n + 255) / 256)), 256, 0, stream>>>(xdbl, dtin, 320, dtr, n);
        }
        launch_gemm(dtin, w_dt + (size_t)l * dtr * din, dtf, Mm, din, dtr);
        {
            long n = (long)Mm * din;
            softplus_bias<<<dim3((unsigned)((n + 255) / 256)), 256, 0, stream>>>(
                dtf, dt_b + (size_t)l * din, din, n);
        }
        mamba_scan<<<dim3(din / 64, B), 256, 0, stream>>>(
            dtf, xc, xdbl, A_log + (size_t)l * din * 128, D_param + (size_t)l * din,
            xz, ybf, T, din);
        launch_gemm(ybf, w_out + (size_t)l * din * D, po, Mm, D, din);
        residual_add<<<dim3(T * D / 256, B), 256, 0, stream>>>(x, po, out, T, D, L, l * T);
    }

    // --- text: gated MLP on 5 groups (shared weights) ---
    for (int g = 0; g < 5; ++g)
        rmsnorm_bf16<<<dim3(Ntok, B), 256, 0, stream>>>(
            x + (size_t)(3 * T + g * Ntok) * D, nt_w + (size_t)g * D,
            nrm + (size_t)g * B * Ntok * D, D, (long)L * D);
    launch_gemm(nrm, w_gb, gbuf, Mt, 4096, D);
    launch_gemm(nrm, w_ub, ubuf, Mt, 4096, D);
    {
        long n = (long)Mt * 4096;
        silu_mul_bf16<<<dim3((unsigned)((n + 255) / 256)), 256, 0, stream>>>(gbuf, ubuf, hbf, n);
    }
    launch_gemm(hbf, w_db, tout, Mt, D, 4096);
    for (int g = 0; g < 5; ++g)
        residual_add<<<dim3(Ntok * D / 256, B), 256, 0, stream>>>(
            x, tout + (size_t)g * B * Ntok * D, out, Ntok, D, L, 3 * T + g * Ntok);
}
